// CausalSelfAttention_13554916786720
// MI455X (gfx1250) — matrix operand order repaired and checked by reading, not compiled, not run on hardware
//
#include <hip/hip_runtime.h>

// ---------------------------------------------------------------------------
// Causal self-attention, B=4, S=2048, D=1024, via CDNA5 WMMA (bf16 in, f32 acc)
// ws layout (bf16): Q[8192][1024] | K[8192][1024] | V[8192][1024] | P[4][2048][2048]
// ---------------------------------------------------------------------------

#define BATCH 4
#define SEQ   2048
#define DIM   1024
#define ROWS  (BATCH * SEQ)          // 8192 flattened rows for projections

typedef unsigned short u16;
typedef __attribute__((ext_vector_type(16))) __bf16 v16bf;
typedef __attribute__((ext_vector_type(8)))  float  v8f;

union Frag { v16bf v; uint4 q[2]; };
union U4U16 { uint4 q; u16 s[8]; };

__device__ __forceinline__ u16 bf16_rn(float f) {
    union { float f; unsigned u; } c; c.f = f;
    unsigned u = c.u;
    unsigned r = u + 0x7FFFu + ((u >> 16) & 1u);   // round-to-nearest-even
    return (u16)(r >> 16);
}

// packed f32x2 -> bf16x2 (v_cvt_pk_bf16_f32 when available, RNE either way)
__device__ __forceinline__ unsigned bf16x2_rn(float lo, float hi) {
#if __has_builtin(__builtin_amdgcn_cvt_pk_bf16_f32)
    typedef __attribute__((ext_vector_type(2))) __bf16 v2bf_t;
    union { v2bf_t v; unsigned u; } c;
    c.v = __builtin_amdgcn_cvt_pk_bf16_f32(lo, hi);
    return c.u;
#else
    return (unsigned)bf16_rn(lo) | ((unsigned)bf16_rn(hi) << 16);
#endif
}

__device__ __forceinline__ float redmax16(float v) {   // reduce within lanes 0-15 / 16-31
    v = fmaxf(v, __shfl_xor(v, 1, 32));
    v = fmaxf(v, __shfl_xor(v, 2, 32));
    v = fmaxf(v, __shfl_xor(v, 4, 32));
    v = fmaxf(v, __shfl_xor(v, 8, 32));
    return v;
}
__device__ __forceinline__ float redsum16(float v) {
    v += __shfl_xor(v, 1, 32);
    v += __shfl_xor(v, 2, 32);
    v += __shfl_xor(v, 4, 32);
    v += __shfl_xor(v, 8, 32);
    return v;
}

// CDNA5 async global->LDS copy (ASYNCcnt-tracked, no VGPR round trip).
// Per-lane: copies 16 bytes from global address vaddr to LDS address vdst.  (CDNA5 ISA)
__device__ __forceinline__ void async_copy_b128(void* lds_dst, const void* gsrc) {
    unsigned ldso = (unsigned)(size_t)lds_dst;              // generic ptr low 32b = LDS offset
    unsigned long long g = (unsigned long long)(size_t)gsrc;
    asm volatile("global_load_async_to_lds_b128 %0, %1, off"
                 :: "v"(ldso), "v"(g) : "memory");
}
__device__ __forceinline__ void wait_async0() {
    asm volatile("s_wait_asynccnt 0" ::: "memory");
}

// ---------------------------------------------------------------------------
// Kernel 1: Q/K/V = x @ W{q,k,v}.  f32 -> bf16 on the fly, bf16 WMMA, f32 acc.
// Block: 256 thr (8 waves). Block tile 64(M) x 128(N). Wave tile 32x32 (2x2 WMMA).
// ---------------------------------------------------------------------------
__global__ __launch_bounds__(256)
void qkv_proj_kernel(const float* __restrict__ x,
                     const float* __restrict__ Wq,
                     const float* __restrict__ Wk,
                     const float* __restrict__ Wv,
                     u16* __restrict__ Qb, u16* __restrict__ Kb, u16* __restrict__ Vb)
{
    const int tid  = threadIdx.x;
    const int m0   = blockIdx.y * 64;
    const int n0   = blockIdx.x * 128;
    const float* W; u16* O;
    if (blockIdx.z == 0)      { W = Wq; O = Qb; }
    else if (blockIdx.z == 1) { W = Wk; O = Kb; }
    else                      { W = Wv; O = Vb; }

    __shared__ u16 As[64 * 32];     // A tile, row-major (m, k), bf16
    __shared__ u16 Bs[128 * 32];    // W tile TRANSPOSED (n, k), bf16

    const int lane = tid & 31;
    const int wave = tid >> 5;
    const int wm   = (wave & 1) * 32;
    const int wn   = (wave >> 1) * 32;
    const int half = lane >> 4;
    const int lr   = lane & 15;

    v8f acc[2][2];
    #pragma unroll
    for (int mi = 0; mi < 2; ++mi)
        #pragma unroll
        for (int ni = 0; ni < 2; ++ni)
            acc[mi][ni] = (v8f){0.f,0.f,0.f,0.f,0.f,0.f,0.f,0.f};

    for (int k0 = 0; k0 < DIM; k0 += 32) {
        // stage A: 64x32 f32 -> bf16  (512 float4, 2 per thread), packed b32 stores
        #pragma unroll
        for (int t = 0; t < 2; ++t) {
            int i   = tid + t * 256;
            int row = i >> 3;
            int c4  = (i & 7) * 4;
            float4 f = *(const float4*)(x + (size_t)(m0 + row) * DIM + k0 + c4);
            unsigned* d32 = (unsigned*)&As[row * 32 + c4];
            d32[0] = bf16x2_rn(f.x, f.y);
            d32[1] = bf16x2_rn(f.z, f.w);
        }
        // stage W transposed: 32x128 f32 -> bf16 (1024 float4, 4 per thread)
        #pragma unroll
        for (int t = 0; t < 4; ++t) {
            int i  = tid + t * 256;
            int kr = i >> 5;
            int c4 = (i & 31) * 4;
            float4 f = *(const float4*)(W + (size_t)(k0 + kr) * DIM + n0 + c4);
            Bs[(c4 + 0) * 32 + kr] = bf16_rn(f.x);
            Bs[(c4 + 1) * 32 + kr] = bf16_rn(f.y);
            Bs[(c4 + 2) * 32 + kr] = bf16_rn(f.z);
            Bs[(c4 + 3) * 32 + kr] = bf16_rn(f.w);
        }
        if (k0 + 32 < DIM)   // warm L2/L0 for the next A tile
            __builtin_prefetch(x + (size_t)(m0 + (tid >> 2)) * DIM + k0 + 32, 0, 1);
        __syncthreads();

        Frag a[2], b[2];
        #pragma unroll
        for (int mi = 0; mi < 2; ++mi) {
            const uint4* p = (const uint4*)&As[(wm + mi * 16 + lr) * 32];
            a[mi].q[0] = p[half];          // K 0-7  / 8-15
            a[mi].q[1] = p[2 + half];      // K 16-23 / 24-31
        }
        #pragma unroll
        for (int ni = 0; ni < 2; ++ni) {
            const uint4* p = (const uint4*)&Bs[(wn + ni * 16 + lr) * 32];
            b[ni].q[0] = p[half];          // K 0-7  / 8-15
            b[ni].q[1] = p[2 + half];      // K 16-23 / 24-31
        }
        #pragma unroll
        for (int mi = 0; mi < 2; ++mi)
            #pragma unroll
            for (int ni = 0; ni < 2; ++ni)
                acc[mi][ni] = __builtin_amdgcn_wmma_f32_16x16x32_bf16(
                    false, a[mi].v, false, b[ni].v, (short)0, acc[mi][ni], false, false);
        __syncthreads();
    }

    #pragma unroll
    for (int mi = 0; mi < 2; ++mi)
        #pragma unroll
        for (int ni = 0; ni < 2; ++ni)
            #pragma unroll
            for (int r = 0; r < 8; ++r) {
                int row = m0 + wm + mi * 16 + r + half * 8;
                int col = n0 + wn + ni * 16 + lr;
                O[(size_t)row * DIM + col] = bf16_rn(acc[mi][ni][r]);
            }
}

// ---------------------------------------------------------------------------
// Kernel 2: P = softmax(causal(Q K^T / 32)), bf16 out.
// Block: 128 thr (4 waves); wave owns 16 query rows; block covers 64 rows.
// Fragments load straight from global (row-major Q and K match the WMMA
// A / B per-lane layouts -> pure global_load_b128, no transpose).
// Two 16-key tiles per d-sweep share one Q fragment (halves Q traffic).
// Pass 1: online max/sum. Pass 2: recompute, write normalized P.
// ---------------------------------------------------------------------------
__global__ __launch_bounds__(128)
void attn_softmax_kernel(const u16* __restrict__ Qb,
                         const u16* __restrict__ Kb,
                         u16* __restrict__ Pb)
{
    const int tid  = threadIdx.x;
    const int wave = tid >> 5;
    const int lane = tid & 31;
    const int half = lane >> 4;
    const int lr   = lane & 15;
    const int qb   = blockIdx.x * 64;
    const int q0   = qb + wave * 16;
    const int bat  = blockIdx.y;

    const u16* Q = Qb + (size_t)bat * SEQ * DIM;
    const u16* K = Kb + (size_t)bat * SEQ * DIM;
    u16*       P = Pb + (size_t)bat * SEQ * SEQ;

    const int ktmax = q0 >> 4;          // diagonal 16-key tile (inclusive)
    const float scale = 0.03125f;       // 1/sqrt(1024)
    const v8f vzero = (v8f){0.f,0.f,0.f,0.f,0.f,0.f,0.f,0.f};

    float m[8], l[8];
    #pragma unroll
    for (int r = 0; r < 8; ++r) { m[r] = -1e30f; l[r] = 0.f; }

    // ---- pass 1: online row max / sum ----
    for (int kt = 0; kt <= ktmax; kt += 2) {
        const int  k0a  = kt * 16;
        const bool has2 = (kt + 1) <= ktmax;
        const int  k0b  = has2 ? k0a + 16 : k0a;      // duplicate tail tile (ignored)
        v8f accs[2] = { vzero, vzero };
        for (int d0 = 0; d0 < DIM; d0 += 32) {
            Frag a, b0, b1;
            const uint4* pq = (const uint4*)(Q + (size_t)(q0 + lr) * DIM + d0);
            a.q[0] = pq[half];  a.q[1] = pq[2 + half];
            const uint4* pk0 = (const uint4*)(K + (size_t)(k0a + lr) * DIM + d0);
            b0.q[0] = pk0[half]; b0.q[1] = pk0[2 + half];
            const uint4* pk1 = (const uint4*)(K + (size_t)(k0b + lr) * DIM + d0);
            b1.q[0] = pk1[half]; b1.q[1] = pk1[2 + half];
            accs[0] = __builtin_amdgcn_wmma_f32_16x16x32_bf16(
                false, a.v, false, b0.v, (short)0, accs[0], false, false);
            accs[1] = __builtin_amdgcn_wmma_f32_16x16x32_bf16(
                false, a.v, false, b1.v, (short)0, accs[1], false, false);
        }
        const int nt = has2 ? 2 : 1;
        for (int t = 0; t < nt; ++t) {
            const int k0 = k0a + t * 16;
            #pragma unroll
            for (int r = 0; r < 8; ++r) {
                int row = q0 + r + half * 8;
                int col = k0 + lr;
                float s = accs[t][r] * scale;
                if (col > row) s = -3.0e38f;            // causal mask
                float tmax = redmax16(s);
                float mnew = fmaxf(m[r], tmax);
                float p    = __expf(s - mnew);
                float rsum = redsum16(p);
                l[r] = l[r] * __expf(m[r] - mnew) + rsum;
                m[r] = mnew;
            }
        }
    }
    float inv[8];
    #pragma unroll
    for (int r = 0; r < 8; ++r) inv[r] = 1.0f / l[r];

    // ---- pass 2: recompute, write normalized P (bf16) ----
    for (int kt = 0; kt <= ktmax; kt += 2) {
        const int  k0a  = kt * 16;
        const bool has2 = (kt + 1) <= ktmax;
        const int  k0b  = has2 ? k0a + 16 : k0a;
        v8f accs[2] = { vzero, vzero };
        for (int d0 = 0; d0 < DIM; d0 += 32) {
            Frag a, b0, b1;
            const uint4* pq = (const uint4*)(Q + (size_t)(q0 + lr) * DIM + d0);
            a.q[0] = pq[half];  a.q[1] = pq[2 + half];
            const uint4* pk0 = (const uint4*)(K + (size_t)(k0a + lr) * DIM + d0);
            b0.q[0] = pk0[half]; b0.q[1] = pk0[2 + half];
            const uint4* pk1 = (const uint4*)(K + (size_t)(k0b + lr) * DIM + d0);
            b1.q[0] = pk1[half]; b1.q[1] = pk1[2 + half];
            accs[0] = __builtin_amdgcn_wmma_f32_16x16x32_bf16(
                false, a.v, false, b0.v, (short)0, accs[0], false, false);
            accs[1] = __builtin_amdgcn_wmma_f32_16x16x32_bf16(
                false, a.v, false, b1.v, (short)0, accs[1], false, false);
        }
        const int nt = has2 ? 2 : 1;
        for (int t = 0; t < nt; ++t) {
            const int k0 = k0a + t * 16;
            #pragma unroll
            for (int r = 0; r < 8; ++r) {
                int row = q0 + r + half * 8;
                int col = k0 + lr;
                float s = accs[t][r] * scale;
                if (col > row) s = -3.0e38f;
                float p = __expf(s - m[r]) * inv[r];
                P[(size_t)row * SEQ + col] = bf16_rn(p);
            }
        }
    }
    // ---- zero-fill tiles above the diagonal that kernel 3 will read ----
    for (int kt = ktmax + 1; kt < ((qb + 64) >> 4); ++kt) {
        const int k0 = kt * 16;
        #pragma unroll
        for (int r = 0; r < 8; ++r) {
            int row = q0 + r + half * 8;
            P[(size_t)row * SEQ + k0 + lr] = (u16)0;
        }
    }
}

// ---------------------------------------------------------------------------
// Kernel 3: out = P @ V (per batch), causal K bound, f32 out.
// 64x128 block tile. P tile staged via CDNA5 async global->LDS DMA
// (global_load_async_to_lds_b128, ASYNCcnt); V transposed through registers.
// ---------------------------------------------------------------------------
__global__ __launch_bounds__(256)
void attn_pv_kernel(const u16* __restrict__ Pb,
                    const u16* __restrict__ Vb,
                    float* __restrict__ out)
{
    const int tid = threadIdx.x;
    const int m0  = blockIdx.y * 64;        // query rows within batch
    const int n0  = blockIdx.x * 128;
    const int bat = blockIdx.z;

    const u16* P = Pb + (size_t)bat * SEQ * SEQ;
    const u16* V = Vb + (size_t)bat * SEQ * DIM;
    float*     O = out + (size_t)bat * SEQ * DIM;

    __shared__ u16 As[64 * 32];     // P tile (q, k)
    __shared__ u16 Bs[128 * 32];    // V tile transposed (n, k)

    const int lane = tid & 31;
    const int wave = tid >> 5;
    const int wm   = (wave & 1) * 32;
    const int wn   = (wave >> 1) * 32;
    const int half = lane >> 4;
    const int lr   = lane & 15;
    const int kend = m0 + 64;       // keys > last row of this tile contribute 0

    v8f acc[2][2];
    #pragma unroll
    for (int mi = 0; mi < 2; ++mi)
        #pragma unroll
        for (int ni = 0; ni < 2; ++ni)
            acc[mi][ni] = (v8f){0.f,0.f,0.f,0.f,0.f,0.f,0.f,0.f};

    // per-thread P-tile slot: 64x32 bf16 = 256 x b128, 1 per thread
    const int prow = tid >> 2;
    const int pc8  = (tid & 3) * 8;

    for (int k0 = 0; k0 < kend; k0 += 32) {
        // stage P tile asynchronously: global -> LDS, no VGPR round trip
        async_copy_b128(&As[prow * 32 + pc8],
                        P + (size_t)(m0 + prow) * SEQ + k0 + pc8);
        // stage V tile transposed: 32x128 bf16 = 512 uint4, 2 per thread
        #pragma unroll
        for (int t = 0; t < 2; ++t) {
            int i  = tid + t * 256;
            int kr = i >> 4;
            int c8 = (i & 15) * 8;
            U4U16 u;
            u.q = *(const uint4*)(V + (size_t)(k0 + kr) * DIM + n0 + c8);
            #pragma unroll
            for (int j = 0; j < 8; ++j)
                Bs[(c8 + j) * 32 + kr] = u.s[j];
        }
        wait_async0();                // our async DMA landed in LDS
        __syncthreads();              // everyone's stores + DMAs visible

        Frag a[2], b[2];
        #pragma unroll
        for (int mi = 0; mi < 2; ++mi) {
            const uint4* p = (const uint4*)&As[(wm + mi * 16 + lr) * 32];
            a[mi].q[0] = p[half]; a[mi].q[1] = p[2 + half];
        }
        #pragma unroll
        for (int ni = 0; ni < 2; ++ni) {
            const uint4* p = (const uint4*)&Bs[(wn + ni * 16 + lr) * 32];
            b[ni].q[0] = p[half]; b[ni].q[1] = p[2 + half];
        }
        #pragma unroll
        for (int mi = 0; mi < 2; ++mi)
            #pragma unroll
            for (int ni = 0; ni < 2; ++ni)
                acc[mi][ni] = __builtin_amdgcn_wmma_f32_16x16x32_bf16(
                    false, a[mi].v, false, b[ni].v, (short)0, acc[mi][ni], false, false);
        __syncthreads();
    }

    #pragma unroll
    for (int mi = 0; mi < 2; ++mi)
        #pragma unroll
        for (int ni = 0; ni < 2; ++ni)
            #pragma unroll
            for (int r = 0; r < 8; ++r) {
                int row = m0 + wm + mi * 16 + r + half * 8;
                int col = n0 + wn + ni * 16 + lr;
                O[(size_t)row * DIM + col] = acc[mi][ni][r];
            }
}

// ---------------------------------------------------------------------------
extern "C" void kernel_launch(void* const* d_in, const int* in_sizes, int n_in,
                              void* d_out, int out_size, void* d_ws, size_t ws_size,
                              hipStream_t stream)
{
    const float* x  = (const float*)d_in[0];
    const float* Wq = (const float*)d_in[1];
    const float* Wk = (const float*)d_in[2];
    const float* Wv = (const float*)d_in[3];

    u16* Qb = (u16*)d_ws;                                  // 16 MB
    u16* Kb = Qb + (size_t)ROWS * DIM;                     // 16 MB
    u16* Vb = Kb + (size_t)ROWS * DIM;                     // 16 MB
    u16* Pb = Vb + (size_t)ROWS * DIM;                     // 32 MB
    float* out = (float*)d_out;

    // 1) Q/K/V projections (bf16 WMMA GEMM), z = {Wq, Wk, Wv}
    qkv_proj_kernel<<<dim3(DIM / 128, ROWS / 64, 3), 256, 0, stream>>>(
        x, Wq, Wk, Wv, Qb, Kb, Vb);

    // 2) causal softmax -> P (bf16)
    attn_softmax_kernel<<<dim3(SEQ / 64, BATCH), 128, 0, stream>>>(Qb, Kb, Pb);

    // 3) out = P @ V (f32)
    attn_pv_kernel<<<dim3(DIM / 128, SEQ / 64, BATCH), 256, 0, stream>>>(Pb, Vb, out);
}
